// FEDformer_65103114273256
// MI455X (gfx1250) — matrix operand order repaired and checked by reading, not compiled, not run on hardware
//
#include <hip/hip_runtime.h>
#include <hip/hip_bf16.h>
#include <math.h>

typedef __bf16 bf16;
typedef __attribute__((ext_vector_type(16))) __bf16 v16bf;
typedef __attribute__((ext_vector_type(8)))  float   v8f;

#define BB    32
#define LSEQ  384
#define DM    512
#define NH    8
#define HD    64
#define DFF   2048
#define MM    64
#define M2    128          /* 2*MM interleaved re/im */
#define CIN   7
#define CMARK 4
#define LABELN 192
#define PREDN  192
#define ROWS  (BB*LSEQ)    /* 12288 */
#define FR    (BB*NH*HD)   /* 16384 fourier rows */
#define TPB   256

/* ------------------------------------------------------------------ */
/* WMMA GEMM: C[M,N] = A[M,K](bf16) * B[K,N](bf16) (+bias) (+GELU)     */
/* tiles: 128x128, BK=32; 256 thr = 8 waves in 4x2; wave does 32x64    */
/* A tile staged via GLOBAL_LOAD_ASYNC_TO_LDS_B128 (ASYNCcnt),         */
/* B tile transposed through VGPRs; LDS double-buffered.               */
/* ------------------------------------------------------------------ */
__global__ __launch_bounds__(256)
void k_gemm_bf16(const bf16* __restrict__ A, const bf16* __restrict__ Bm,
                 const float* __restrict__ bias, float* __restrict__ C,
                 int M, int N, int K, int act)
{
  __shared__ __align__(16) __bf16 As[2][128][32];
  __shared__ __align__(16) __bf16 Bst[2][128][40];   /* transposed, padded */

  const int tid  = threadIdx.x;
  const int lane = tid & 31;
  const int wave = tid >> 5;
  const int wm = wave >> 1;          /* 0..3 : 32-row strip   */
  const int wn = wave &  1;          /* 0..1 : 64-col strip   */
  const int lr = lane & 15;
  const int lg = lane >> 4;
  const int bm = blockIdx.y * 128;
  const int bn = blockIdx.x * 128;

  v8f acc[2][4];
  #pragma unroll
  for (int i = 0; i < 2; ++i)
    #pragma unroll
    for (int j = 0; j < 4; ++j) { v8f z = {}; acc[i][j] = z; }

  const int arow = tid >> 1;         /* 0..127 */
  const int aseg = (tid & 1) * 16;   /* 0 / 16 bf16 */
  const int bkr  = tid >> 3;         /* 0..31  */
  const int bn0  = (tid & 7) * 16;   /* 0..112 */

  const bf16* aptr = A + (size_t)(bm + arow) * K + aseg;
  const bf16* bptr = Bm + (size_t)bkr * N + bn + bn0;

  auto load_tile = [&](int buf, int k0) {
    /* A: 32B per thread straight into LDS via async copy (offset applies
       to both the LDS and global address per ISA 15.18 semantics) */
    unsigned lda = (unsigned)(size_t)&As[buf][arow][aseg];
    unsigned long long ga = (unsigned long long)(size_t)(aptr + k0);
    asm volatile("global_load_async_to_lds_b128 %0, %1, off"
                 :: "v"(lda), "v"(ga) : "memory");
    asm volatile("global_load_async_to_lds_b128 %0, %1, off offset:16"
                 :: "v"(lda), "v"(ga) : "memory");
    /* B: 32B per thread, written transposed */
    const uint4* bs = reinterpret_cast<const uint4*>(bptr + (size_t)k0 * N);
    uint4 b0 = bs[0], b1 = bs[1];
    const __bf16* bb0 = reinterpret_cast<const __bf16*>(&b0);
    const __bf16* bb1 = reinterpret_cast<const __bf16*>(&b1);
    #pragma unroll
    for (int j = 0; j < 8; ++j) {
      Bst[buf][bn0 + j][bkr]     = bb0[j];
      Bst[buf][bn0 + 8 + j][bkr] = bb1[j];
    }
  };

  union FragU { uint4 u[2]; v16bf v; };

  load_tile(0, 0);
  const int nsteps = K >> 5;
  for (int it = 0; it < nsteps; ++it) {
    const int cur = it & 1;
    asm volatile("s_wait_asynccnt 0x0" ::: "memory");
    __syncthreads();                          /* tile[cur] visible to all */
    if (it + 1 < nsteps) load_tile(cur ^ 1, (it + 1) << 5);

    /* A fragment: lane (m=lr, half lg): K = lg*8..+7 then 16+lg*8..+7 */
    v16bf af[2], bfr[4];
    #pragma unroll
    for (int mt = 0; mt < 2; ++mt) {
      FragU f;
      const __bf16* arp = &As[cur][wm * 32 + mt * 16 + lr][0];
      f.u[0] = *reinterpret_cast<const uint4*>(arp + lg * 8);
      f.u[1] = *reinterpret_cast<const uint4*>(arp + 16 + lg * 8);
      af[mt] = f.v;
    }
    /* B fragment: lane (n=lr, half lg): K = lg*8..+7 then 16+lg*8..+7 */
    #pragma unroll
    for (int nt = 0; nt < 4; ++nt) {
      FragU f;
      const __bf16* brp = &Bst[cur][wn * 64 + nt * 16 + lr][0];
      f.u[0] = *reinterpret_cast<const uint4*>(brp + lg * 8);
      f.u[1] = *reinterpret_cast<const uint4*>(brp + 16 + lg * 8);
      bfr[nt] = f.v;
    }
    #pragma unroll
    for (int mt = 0; mt < 2; ++mt)
      #pragma unroll
      for (int nt = 0; nt < 4; ++nt)
        acc[mt][nt] = __builtin_amdgcn_wmma_f32_16x16x32_bf16(
            false, af[mt], false, bfr[nt], (short)0, acc[mt][nt], false, false);
  }

  /* store: C/D layout: VGPR r -> M = r + 8*lg ; lane lr -> N */
  #pragma unroll
  for (int mt = 0; mt < 2; ++mt) {
    #pragma unroll
    for (int nt = 0; nt < 4; ++nt) {
      int col = bn + wn * 64 + nt * 16 + lr;
      float bia = bias ? bias[col] : 0.f;
      #pragma unroll
      for (int r = 0; r < 8; ++r) {
        int row = bm + wm * 32 + mt * 16 + lg * 8 + r;
        float v = acc[mt][nt][r] + bia;
        if (act == 1) v = 0.5f * v * (1.f + erff(v * 0.70710678118654752f));
        C[(size_t)row * N + col] = v;
      }
    }
  }
}

/* ---------------------- elementwise / small kernels ----------------- */

__global__ void k_f32_bf16(const float* __restrict__ s, bf16* __restrict__ d, long n)
{
  long i = blockIdx.x * (long)blockDim.x + threadIdx.x;
  if (i < n) d[i] = (bf16)s[i];
}

/* w[R,C] (f32, row-major) -> dst[C,R] (bf16) */
__global__ void k_w_transpose_bf16(const float* __restrict__ w, bf16* __restrict__ d, int R, int Cc)
{
  long i = blockIdx.x * (long)blockDim.x + threadIdx.x;
  if (i >= (long)R * Cc) return;
  int r = (int)(i / Cc), c = (int)(i % Cc);
  d[(size_t)c * R + r] = (bf16)w[i];
}

/* DataEmbedding_wo_pos: circular conv (K=3) + mark @ temp_w */
__global__ void k_embed(const float* __restrict__ x, const float* __restrict__ conv,
                        const float* __restrict__ mark, const float* __restrict__ tw,
                        float* __restrict__ out, int Ci)
{
  long i = blockIdx.x * (long)blockDim.x + threadIdx.x;
  if (i >= (long)BB * LSEQ * DM) return;
  int d = (int)(i % DM); long t = i / DM;
  int l = (int)(t % LSEQ); int b = (int)(t / LSEQ);
  float acc = 0.f;
  for (int ci = 0; ci < Ci; ++ci) {
    #pragma unroll
    for (int kk = 0; kk < 3; ++kk) {
      int ll = l + kk - 1; ll = (ll + LSEQ) % LSEQ;
      acc += x[((size_t)b * LSEQ + ll) * Ci + ci] * conv[((size_t)d * Ci + ci) * 3 + kk];
    }
  }
  #pragma unroll
  for (int j = 0; j < CMARK; ++j)
    acc += mark[((size_t)b * LSEQ + l) * CMARK + j] * tw[(size_t)j * DM + d];
  out[i] = acc;
}

/* seasonal/trend decomposition of (a+c); trend optional (write or accumulate) */
__global__ void k_add_decomp(const float* __restrict__ a, const float* __restrict__ c,
                             float* __restrict__ seas, float* __restrict__ trend,
                             int accum, int L, int D)
{
  long i = blockIdx.x * (long)blockDim.x + threadIdx.x;
  if (i >= (long)BB * L * D) return;
  int d = (int)(i % D); long t = i / D;
  int l = (int)(t % L); int b = (int)(t / L);
  const float* ab = a + (size_t)b * L * D + d;
  const float* cb = c + (size_t)b * L * D + d;
  float s = 0.f;
  #pragma unroll
  for (int j = -12; j <= 12; ++j) {
    int ll = l + j; ll = ll < 0 ? 0 : (ll >= L ? L - 1 : ll);
    s += ab[(size_t)ll * D] + cb[(size_t)ll * D];
  }
  float m = s * (1.f / 25.f);
  float xc = ab[(size_t)l * D] + cb[(size_t)l * D];
  seas[i] = xc - m;
  if (trend) trend[i] = accum ? trend[i] + m : m;
}

/* decomposition of a single small tensor [B,L,C] */
__global__ void k_decomp_small(const float* __restrict__ x, float* __restrict__ seas,
                               float* __restrict__ trend, int L, int C)
{
  long i = blockIdx.x * (long)blockDim.x + threadIdx.x;
  if (i >= (long)BB * L * C) return;
  int c = (int)(i % C); long t = i / C;
  int l = (int)(t % L); int b = (int)(t / L);
  const float* xb = x + (size_t)b * L * C + c;
  float s = 0.f;
  #pragma unroll
  for (int j = -12; j <= 12; ++j) {
    int ll = l + j; ll = ll < 0 ? 0 : (ll >= L ? L - 1 : ll);
    s += xb[(size_t)ll * C];
  }
  float m = s * (1.f / 25.f);
  seas[i] = xb[(size_t)l * C] - m;
  trend[i] = m;
}

__global__ void k_chan_mean(const float* __restrict__ x, float* __restrict__ m, int L, int C)
{
  int i = blockIdx.x * blockDim.x + threadIdx.x;
  if (i >= BB * C) return;
  int b = i / C, c = i % C;
  float s = 0.f;
  for (int l = 0; l < L; ++l) s += x[((size_t)b * L + l) * C + c];
  m[i] = s / L;
}

/* build decoder seasonal_init (zero-padded) and trend_init (mean-extended) */
__global__ void k_build_dec(const float* __restrict__ seas, const float* __restrict__ trend,
                            const float* __restrict__ cm,
                            float* __restrict__ seas_pad, float* __restrict__ tr0, int C)
{
  long i = blockIdx.x * (long)blockDim.x + threadIdx.x;
  if (i >= (long)BB * LSEQ * C) return;
  int c = (int)(i % C); long t = i / C;
  int l = (int)(t % LSEQ); int b = (int)(t / LSEQ);
  if (l < LABELN) {
    seas_pad[i] = seas[((size_t)b * LSEQ + (LSEQ - LABELN + l)) * C + c];
    tr0[i]      = trend[((size_t)b * LSEQ + (LSEQ - LABELN + l)) * C + c];
  } else {
    seas_pad[i] = 0.f;
    tr0[i]      = cm[b * C + c];
  }
}

/* [B,L,512] f32 -> bf16 [B*H*E, L] (head-major transpose) */
__global__ void k_pack_heads(const float* __restrict__ q, bf16* __restrict__ d)
{
  long i = blockIdx.x * (long)blockDim.x + threadIdx.x;
  if (i >= (long)FR * LSEQ) return;
  int l = (int)(i % LSEQ); long r = i / LSEQ;
  int e = (int)(r % HD); int h = (int)((r / HD) % NH); int b = (int)(r / (HD * NH));
  d[i] = (bf16)q[((size_t)b * LSEQ + l) * DM + h * HD + e];
}

/* DFT basis F[L, 2M]: col 2m = cos(wlm), 2m+1 = -sin(wlm)  (rfft, first M modes) */
__global__ void k_dft_basis(bf16* __restrict__ F)
{
  long i = blockIdx.x * (long)blockDim.x + threadIdx.x;
  if (i >= (long)LSEQ * MM) return;
  int m = (int)(i % MM), l = (int)(i / MM);
  float ang = (6.28318530717958647f / LSEQ) * (float)l * (float)m;
  F[(size_t)l * M2 + 2 * m]     = (bf16)cosf(ang);
  F[(size_t)l * M2 + 2 * m + 1] = (bf16)(-sinf(ang));
}

/* iDFT basis G[2M, L]: x_l = (1/L) * (re0 + 2*sum_{m>0} (re_m cos - im_m sin)) */
__global__ void k_idft_basis(bf16* __restrict__ G)
{
  long i = blockIdx.x * (long)blockDim.x + threadIdx.x;
  if (i >= (long)MM * LSEQ) return;
  int l = (int)(i % LSEQ), m = (int)(i / LSEQ);
  float sc = (m == 0 ? 1.f : 2.f) / (float)LSEQ;
  float ang = (6.28318530717958647f / LSEQ) * (float)l * (float)m;
  G[(size_t)(2 * m) * LSEQ + l]     = (bf16)(sc * cosf(ang));
  G[(size_t)(2 * m + 1) * LSEQ + l] = (bf16)(-sc * sinf(ang));
}

/* out[b,h,o,m] = scale * sum_i sel[b,h,i,m] * (wr+j*wi)[h,i,o,m]  (complex, interleaved) */
__global__ void k_modemix(const float* __restrict__ sel, const float* __restrict__ wr,
                          const float* __restrict__ wi, float* __restrict__ out, float scale)
{
  long i = blockIdx.x * (long)blockDim.x + threadIdx.x;
  if (i >= (long)BB * NH * HD * MM) return;
  int m = (int)(i % MM); long t = i / MM;
  int o = (int)(t % HD); int h = (int)((t / HD) % NH); int b = (int)(t / (HD * NH));
  float ar = 0.f, ai = 0.f;
  for (int e = 0; e < HD; ++e) {
    size_t si = (((size_t)(b * NH + h) * HD) + e) * M2 + 2 * m;
    float sr = sel[si], sim = sel[si + 1];
    size_t wiix = (((size_t)h * HD + e) * HD + o) * MM + m;
    float wrv = wr[wiix], wiv = wi[wiix];
    ar += sr * wrv - sim * wiv;
    ai += sr * wiv + sim * wrv;
  }
  size_t oi = (((size_t)(b * NH + h) * HD) + o) * M2 + 2 * m;
  out[oi] = ar * scale;
  out[oi + 1] = ai * scale;
}

/* qk[b,h,x,y] = tanh( sum_e qf[b,h,e,x] * kf[b,h,e,y] )   (complex tanh) */
__global__ void k_qk_tanh(const float* __restrict__ qf, const float* __restrict__ kf,
                          float* __restrict__ qk)
{
  long i = blockIdx.x * (long)blockDim.x + threadIdx.x;
  if (i >= (long)BB * NH * MM * MM) return;
  int y = (int)(i % MM); long t = i / MM;
  int x = (int)(t % MM); int h = (int)((t / MM) % NH); int b = (int)(t / (MM * NH));
  float ar = 0.f, ai = 0.f;
  for (int e = 0; e < HD; ++e) {
    size_t base = (((size_t)(b * NH + h) * HD) + e) * M2;
    float qr = qf[base + 2 * x], qi = qf[base + 2 * x + 1];
    float kr = kf[base + 2 * y], ki = kf[base + 2 * y + 1];
    ar += qr * kr - qi * ki;
    ai += qr * ki + qi * kr;
  }
  float ta = tanhf(ar), tb = tanf(ai);
  float den = 1.f + ta * ta * tb * tb;
  float rr = ta * (1.f + tb * tb) / den;
  float ri = tb * (1.f - ta * ta) / den;
  size_t oi = (((size_t)(b * NH + h) * MM) + x) * M2 + 2 * y;
  qk[oi] = rr;
  qk[oi + 1] = ri;
}

/* qkv[b,h,e,x] = sum_y qk[b,h,x,y] * kf[b,h,e,y] */
__global__ void k_qkv(const float* __restrict__ qk, const float* __restrict__ kf,
                      float* __restrict__ qkv)
{
  long i = blockIdx.x * (long)blockDim.x + threadIdx.x;
  if (i >= (long)BB * NH * HD * MM) return;
  int x = (int)(i % MM); long t = i / MM;
  int e = (int)(t % HD); int h = (int)((t / HD) % NH); int b = (int)(t / (HD * NH));
  float ar = 0.f, ai = 0.f;
  size_t qbase = (((size_t)(b * NH + h) * MM) + x) * M2;
  size_t kbase = (((size_t)(b * NH + h) * HD) + e) * M2;
  for (int y = 0; y < MM; ++y) {
    float qr = qk[qbase + 2 * y], qi = qk[qbase + 2 * y + 1];
    float kr = kf[kbase + 2 * y], ki = kf[kbase + 2 * y + 1];
    ar += qr * kr - qi * ki;
    ai += qr * ki + qi * kr;
  }
  size_t oi = kbase + 2 * x;
  qkv[oi] = ar;
  qkv[oi + 1] = ai;
}

/* my_Layernorm part 1: plain LN over D (=512), block per row, 256 threads */
__global__ __launch_bounds__(256)
void k_layernorm(const float* __restrict__ x, const float* __restrict__ g,
                 const float* __restrict__ be, float* __restrict__ y)
{
  int row = blockIdx.x;
  const float* xr = x + (size_t)row * DM;
  float* yr = y + (size_t)row * DM;
  int t = threadIdx.x;
  __shared__ float sd[256];
  float v0 = xr[t], v1 = xr[t + 256];
  sd[t] = v0 + v1;
  __syncthreads();
  for (int s = 128; s > 0; s >>= 1) { if (t < s) sd[t] += sd[t + s]; __syncthreads(); }
  float mu = sd[0] / DM;
  __syncthreads();
  float d0 = v0 - mu, d1 = v1 - mu;
  sd[t] = d0 * d0 + d1 * d1;
  __syncthreads();
  for (int s = 128; s > 0; s >>= 1) { if (t < s) sd[t] += sd[t + s]; __syncthreads(); }
  float rstd = rsqrtf(sd[0] / DM + 1e-5f);
  yr[t]       = d0 * rstd * g[t]       + be[t];
  yr[t + 256] = d1 * rstd * g[t + 256] + be[t + 256];
}

/* my_Layernorm part 2: subtract per-(b,d) mean over the sequence axis */
__global__ void k_seqmean_sub(float* __restrict__ y, int L, int D)
{
  int i = blockIdx.x * blockDim.x + threadIdx.x;
  if (i >= BB * D) return;
  int b = i / D, d = i % D;
  float* p = y + (size_t)b * L * D + d;
  float s = 0.f;
  for (int l = 0; l < L; ++l) s += p[(size_t)l * D];
  s /= L;
  for (int l = 0; l < L; ++l) p[(size_t)l * D] -= s;
}

/* trend projection: out = tr0 + circular_conv(t512, w[7,512,3]) */
__global__ void k_trend_conv(const float* __restrict__ t512, const float* __restrict__ w,
                             const float* __restrict__ tr0, float* __restrict__ out)
{
  long i = blockIdx.x * (long)blockDim.x + threadIdx.x;
  if (i >= (long)BB * LSEQ * CIN) return;
  int c = (int)(i % CIN); long t = i / CIN;
  int l = (int)(t % LSEQ); int b = (int)(t / LSEQ);
  float acc = tr0[i];
  for (int d = 0; d < DM; ++d) {
    #pragma unroll
    for (int kk = 0; kk < 3; ++kk) {
      int ll = l + kk - 1; ll = (ll + LSEQ) % LSEQ;
      acc += t512[((size_t)b * LSEQ + ll) * DM + d] * w[((size_t)c * DM + d) * 3 + kk];
    }
  }
  out[i] = acc;
}

/* final: out[b,p,c] = trend[b,192+p,c] + dec_out[b,192+p,:] @ proj.w + proj.b */
__global__ void k_final(const float* __restrict__ x, const float* __restrict__ pw,
                        const float* __restrict__ pb, const float* __restrict__ trend,
                        float* __restrict__ out)
{
  long i = blockIdx.x * (long)blockDim.x + threadIdx.x;
  if (i >= (long)BB * PREDN * CIN) return;
  int c = (int)(i % CIN); long t = i / CIN;
  int p = (int)(t % PREDN); int b = (int)(t / PREDN);
  int l = LABELN + p;
  float acc = pb[c] + trend[((size_t)b * LSEQ + l) * CIN + c];
  for (int d = 0; d < DM; ++d)
    acc += x[((size_t)b * LSEQ + l) * DM + d] * pw[(size_t)d * CIN + c];
  out[i] = acc;
}

/* ------------------------------ host side --------------------------- */

static inline dim3 g1(long n) { return dim3((unsigned)((n + TPB - 1) / TPB)); }

struct Bufs {
  float *F0, *F1, *F2, *F4, *F5, *T512;
  bf16  *ABF, *BBF;
  float *QF, *KF, *QK, *QKV, *SELO;
  float *SF, *TF, *CM, *SEAS, *TR0, *TROUT;
};

static void gemm(const bf16* A, const bf16* Bm, const float* bias, float* C,
                 int M, int N, int K, int act, hipStream_t s)
{
  dim3 grid(N / 128, M / 128);
  k_gemm_bf16<<<grid, dim3(256), 0, s>>>(A, Bm, bias, C, M, N, K, act);
}

static void to_bf16(const float* src, bf16* dst, long n, hipStream_t s)
{
  k_f32_bf16<<<g1(n), TPB, 0, s>>>(src, dst, n);
}

/* FEB-f self attention block (only q is used by the fourier corr) */
static void fourier_self_block(const float* x,
                               const float* qw, const float* qb,
                               const float* ow, const float* ob,
                               const float* wr, const float* wi,
                               const Bufs& W, float* out, hipStream_t s)
{
  to_bf16(x, W.ABF, (long)ROWS * DM, s);
  to_bf16(qw, W.BBF, (long)DM * DM, s);
  gemm(W.ABF, W.BBF, qb, W.F1, ROWS, DM, DM, 0, s);            /* q proj */
  k_pack_heads<<<g1((long)FR * LSEQ), TPB, 0, s>>>(W.F1, W.ABF);
  k_dft_basis<<<g1((long)LSEQ * MM), TPB, 0, s>>>(W.BBF);
  gemm(W.ABF, W.BBF, nullptr, W.QF, FR, M2, LSEQ, 0, s);       /* rfft (64 modes) */
  k_modemix<<<g1((long)BB * NH * HD * MM), TPB, 0, s>>>(W.QF, wr, wi, W.SELO, 1.0f);
  to_bf16(W.SELO, W.ABF, (long)FR * M2, s);
  k_idft_basis<<<g1((long)MM * LSEQ), TPB, 0, s>>>(W.BBF);
  gemm(W.ABF, W.BBF, nullptr, W.F1, FR, LSEQ, M2, 0, s);       /* irfft */
  /* [B,H,E,L] contiguous == [B,L',512] flat view (faithful reshape) */
  to_bf16(W.F1, W.ABF, (long)ROWS * DM, s);
  to_bf16(ow, W.BBF, (long)DM * DM, s);
  gemm(W.ABF, W.BBF, ob, out, ROWS, DM, DM, 0, s);             /* o proj */
}

/* FEA-f cross attention block (q from decoder, k from encoder; v unused) */
static void fourier_cross_block(const float* xq, const float* xkv,
                                const float* qw, const float* qb,
                                const float* kw, const float* kb,
                                const float* ow, const float* ob,
                                const float* wr, const float* wi,
                                const Bufs& W, float* out, hipStream_t s)
{
  to_bf16(xq, W.ABF, (long)ROWS * DM, s);
  to_bf16(qw, W.BBF, (long)DM * DM, s);
  gemm(W.ABF, W.BBF, qb, W.F1, ROWS, DM, DM, 0, s);
  k_pack_heads<<<g1((long)FR * LSEQ), TPB, 0, s>>>(W.F1, W.ABF);
  k_dft_basis<<<g1((long)LSEQ * MM), TPB, 0, s>>>(W.BBF);
  gemm(W.ABF, W.BBF, nullptr, W.QF, FR, M2, LSEQ, 0, s);       /* qf */

  to_bf16(xkv, W.ABF, (long)ROWS * DM, s);
  to_bf16(kw, W.BBF, (long)DM * DM, s);
  gemm(W.ABF, W.BBF, kb, W.F1, ROWS, DM, DM, 0, s);
  k_pack_heads<<<g1((long)FR * LSEQ), TPB, 0, s>>>(W.F1, W.ABF);
  k_dft_basis<<<g1((long)LSEQ * MM), TPB, 0, s>>>(W.BBF);
  gemm(W.ABF, W.BBF, nullptr, W.KF, FR, M2, LSEQ, 0, s);       /* kf */

  k_qk_tanh<<<g1((long)BB * NH * MM * MM), TPB, 0, s>>>(W.QF, W.KF, W.QK);
  k_qkv<<<g1((long)BB * NH * HD * MM), TPB, 0, s>>>(W.QK, W.KF, W.QKV);
  k_modemix<<<g1((long)BB * NH * HD * MM), TPB, 0, s>>>(W.QKV, wr, wi, W.SELO,
                                                        1.0f / (512.f * 512.f));
  to_bf16(W.SELO, W.ABF, (long)FR * M2, s);
  k_idft_basis<<<g1((long)MM * LSEQ), TPB, 0, s>>>(W.BBF);
  gemm(W.ABF, W.BBF, nullptr, W.F1, FR, LSEQ, M2, 0, s);
  to_bf16(W.F1, W.ABF, (long)ROWS * DM, s);
  to_bf16(ow, W.BBF, (long)DM * DM, s);
  gemm(W.ABF, W.BBF, ob, out, ROWS, DM, DM, 0, s);
}

static void ffn_block(const float* x, const float* conv1, const float* conv2,
                      const Bufs& W, float* out, hipStream_t s)
{
  to_bf16(x, W.ABF, (long)ROWS * DM, s);
  k_w_transpose_bf16<<<g1((long)DFF * DM), TPB, 0, s>>>(conv1, W.BBF, DFF, DM);
  gemm(W.ABF, W.BBF, nullptr, W.F1, ROWS, DFF, DM, 1, s);      /* gelu */
  to_bf16(W.F1, W.ABF, (long)ROWS * DFF, s);
  k_w_transpose_bf16<<<g1((long)DM * DFF), TPB, 0, s>>>(conv2, W.BBF, DM, DFF);
  gemm(W.ABF, W.BBF, nullptr, out, ROWS, DM, DFF, 0, s);
}

/* ---- param leaf indices (jax pytree: dict keys sorted, lists in order) ---- */
enum {
  IN_XENC = 0, IN_MARK_ENC = 1, IN_XDEC = 2, IN_MARK_DEC = 3,
  P_DEC_EMB_CONV = 4, P_DEC_EMB_TW = 5,
  DL_CONV1 = 6, DL_CONV2 = 7,
  DL_CA_KB = 8, DL_CA_KW = 9, DL_CA_OB = 10, DL_CA_OW = 11,
  DL_CA_QB = 12, DL_CA_QW = 13, /* v unused */
  DL_CF_WI = 16, DL_CF_WR = 17,
  DL_SA_OB = 20, DL_SA_OW = 21, DL_SA_QB = 22, DL_SA_QW = 23,
  DL_SF_WI = 26, DL_SF_WR = 27,
  DL_TREND = 28,
  P_DEC_NORM_B = 29, P_DEC_NORM_G = 30,
  P_ENC_EMB_CONV = 31, P_ENC_EMB_TW = 32,
  EL_BASE = 33, EL_STRIDE = 12,
  /* per enc layer: +0 KB +1 KW +2 OB +3 OW +4 QB +5 QW +6 VB +7 VW +8 CONV1 +9 CONV2 +10 WI +11 WR */
  P_ENC_NORM_B = 57, P_ENC_NORM_G = 58,
  P_PROJ_B = 59, P_PROJ_W = 60
};

extern "C" void kernel_launch(void* const* d_in, const int* in_sizes, int n_in,
                              void* d_out, int out_size, void* d_ws, size_t ws_size,
                              hipStream_t stream)
{
  (void)in_sizes; (void)n_in; (void)out_size; (void)ws_size;
  const float* const* in = (const float* const*)d_in;
  float* out = (float*)d_out;

  /* deterministic workspace carve-up */
  char* base = (char*)d_ws;
  size_t off = 0;
  auto carve = [&](size_t bytes) -> void* {
    void* p = base + off;
    off += (bytes + 255) & ~(size_t)255;
    return p;
  };
  Bufs W;
  W.F0   = (float*)carve((size_t)ROWS * DM * 4);
  W.F1   = (float*)carve((size_t)ROWS * DFF * 4);
  W.F2   = (float*)carve((size_t)ROWS * DM * 4);
  W.F4   = (float*)carve((size_t)ROWS * DM * 4);
  W.F5   = (float*)carve((size_t)ROWS * DM * 4);
  W.T512 = (float*)carve((size_t)ROWS * DM * 4);
  W.ABF  = (bf16*)carve((size_t)ROWS * DFF * 2);
  W.BBF  = (bf16*)carve((size_t)DFF * DM * 2);
  W.QF   = (float*)carve((size_t)FR * M2 * 4);
  W.KF   = (float*)carve((size_t)FR * M2 * 4);
  W.QK   = (float*)carve((size_t)FR * M2 * 4);
  W.QKV  = (float*)carve((size_t)FR * M2 * 4);
  W.SELO = (float*)carve((size_t)FR * M2 * 4);
  W.SF   = (float*)carve((size_t)BB * LSEQ * CIN * 4);
  W.TF   = (float*)carve((size_t)BB * LSEQ * CIN * 4);
  W.CM   = (float*)carve((size_t)BB * CIN * 4);
  W.SEAS = (float*)carve((size_t)BB * LSEQ * CIN * 4);
  W.TR0  = (float*)carve((size_t)BB * LSEQ * CIN * 4);
  W.TROUT= (float*)carve((size_t)BB * LSEQ * CIN * 4);

  /* ---------------- encoder ---------------- */
  k_embed<<<g1((long)BB * LSEQ * DM), TPB, 0, stream>>>(
      in[IN_XENC], in[P_ENC_EMB_CONV], in[IN_MARK_ENC], in[P_ENC_EMB_TW], W.F0, CIN);

  for (int layer = 0; layer < 2; ++layer) {
    int p = EL_BASE + layer * EL_STRIDE;
    /* attention (FEB-f uses q only) + residual decomp */
    fourier_self_block(W.F0, in[p + 5], in[p + 4], in[p + 3], in[p + 2],
                       in[p + 11], in[p + 10], W, W.F2, stream);
    k_add_decomp<<<g1((long)ROWS * DM), TPB, 0, stream>>>(
        W.F0, W.F2, W.F0, (float*)nullptr, 0, LSEQ, DM);
    /* FFN + residual decomp */
    ffn_block(W.F0, in[p + 8], in[p + 9], W, W.F2, stream);
    k_add_decomp<<<g1((long)ROWS * DM), TPB, 0, stream>>>(
        W.F0, W.F2, W.F0, (float*)nullptr, 0, LSEQ, DM);
  }
  k_layernorm<<<dim3(ROWS), dim3(256), 0, stream>>>(
      W.F0, in[P_ENC_NORM_G], in[P_ENC_NORM_B], W.F4);
  k_seqmean_sub<<<g1((long)BB * DM), TPB, 0, stream>>>(W.F4, LSEQ, DM);

  /* ---------------- decoder init ---------------- */
  k_decomp_small<<<g1((long)BB * LSEQ * CIN), TPB, 0, stream>>>(in[IN_XENC], W.SF, W.TF, LSEQ, CIN);
  k_chan_mean<<<g1((long)BB * CIN), TPB, 0, stream>>>(in[IN_XENC], W.CM, LSEQ, CIN);
  k_build_dec<<<g1((long)BB * LSEQ * CIN), TPB, 0, stream>>>(W.SF, W.TF, W.CM, W.SEAS, W.TR0, CIN);
  k_embed<<<g1((long)BB * LSEQ * DM), TPB, 0, stream>>>(
      W.SEAS, in[P_DEC_EMB_CONV], in[IN_MARK_DEC], in[P_DEC_EMB_TW], W.F5, CIN);

  /* ---------------- decoder layer ---------------- */
  /* self attention + decomp (init trend sum) */
  fourier_self_block(W.F5, in[DL_SA_QW], in[DL_SA_QB], in[DL_SA_OW], in[DL_SA_OB],
                     in[DL_SF_WR], in[DL_SF_WI], W, W.F2, stream);
  k_add_decomp<<<g1((long)ROWS * DM), TPB, 0, stream>>>(W.F5, W.F2, W.F5, W.T512, 0, LSEQ, DM);
  /* cross attention + decomp (accumulate trend) */
  fourier_cross_block(W.F5, W.F4, in[DL_CA_QW], in[DL_CA_QB], in[DL_CA_KW], in[DL_CA_KB],
                      in[DL_CA_OW], in[DL_CA_OB], in[DL_CF_WR], in[DL_CF_WI],
                      W, W.F2, stream);
  k_add_decomp<<<g1((long)ROWS * DM), TPB, 0, stream>>>(W.F5, W.F2, W.F5, W.T512, 1, LSEQ, DM);
  /* FFN + decomp (accumulate trend) */
  ffn_block(W.F5, in[DL_CONV1], in[DL_CONV2], W, W.F2, stream);
  k_add_decomp<<<g1((long)ROWS * DM), TPB, 0, stream>>>(W.F5, W.F2, W.F5, W.T512, 1, LSEQ, DM);
  /* trend projection */
  k_trend_conv<<<g1((long)BB * LSEQ * CIN), TPB, 0, stream>>>(W.T512, in[DL_TREND], W.TR0, W.TROUT);

  /* decoder norm + final projection */
  k_layernorm<<<dim3(ROWS), dim3(256), 0, stream>>>(
      W.F5, in[P_DEC_NORM_G], in[P_DEC_NORM_B], W.F0);
  k_seqmean_sub<<<g1((long)BB * DM), TPB, 0, stream>>>(W.F0, LSEQ, DM);
  k_final<<<g1((long)BB * PREDN * CIN), TPB, 0, stream>>>(
      W.F0, in[P_PROJ_W], in[P_PROJ_B], W.TROUT, out);
}
